// AsyncLSTM_1297080124026
// MI455X (gfx1250) — compile-verified, matrix operand order checked by reading, not run on hardware
//
#include <hip/hip_runtime.h>
#include <hip/hip_bf16.h>

typedef __attribute__((ext_vector_type(16))) __bf16 v16bf;
typedef __attribute__((ext_vector_type(8)))  __bf16 v8bf;
typedef __attribute__((ext_vector_type(8)))  float  v8f;

// ---------------- problem dims ----------------
constexpr int cB  = 256, cT = 128;
constexpr int cDL = 300, cDA = 74;
constexpr int cHL = 512, cHA = 128, cMEM = 128;
constexpr int cKL = 832;    // 320 (x_l padded) + 512 (h_l)
constexpr int cKA = 224;    // 96  (x_a padded) + 128 (h_a)
constexpr int cCSW = 1408;  // cStar 1280 + mem 128
constexpr int NT = 128;     // 4 waves of 32
constexpr int NWAVE = 4;

// ---------------- packed-weight workspace layout (bf16 elements) -------------
// tile = 16(N) x 32(K) bf16 = 512 elements; lane l owns 16 contiguous elems.
constexpr size_t SZ_WL   = (size_t)128 * 26 * 512;  // [2048,832]
constexpr size_t SZ_WA   = (size_t)32  * 7  * 512;  // [512,224]
constexpr size_t SZ_A1W1 = (size_t)8   * 40 * 512;  // [128,1280]
constexpr size_t SZ_A1W2 = (size_t)80  * 4  * 512;  // [1280,128]
constexpr size_t SZ_A2W1 = (size_t)16  * 40 * 512;  // [256,1280]
constexpr size_t SZ_A2W2 = (size_t)8   * 8  * 512;  // [128,256]
constexpr size_t SZ_GW1  = (size_t)4   * 44 * 512;  // [64,1408]
constexpr size_t SZ_GW2  = (size_t)8   * 2  * 512;  // [128,64]
constexpr size_t SZ_OW1  = (size_t)4   * 24 * 512;  // [64,768]

constexpr size_t OFF_WL   = 0;
constexpr size_t OFF_WA   = OFF_WL   + SZ_WL;
constexpr size_t OFF_A1W1 = OFF_WA   + SZ_WA;
constexpr size_t OFF_A1W2 = OFF_A1W1 + SZ_A1W1;
constexpr size_t OFF_A2W1 = OFF_A1W2 + SZ_A1W2;
constexpr size_t OFF_A2W2 = OFF_A2W1 + SZ_A2W1;
constexpr size_t OFF_G1W1 = OFF_A2W2 + SZ_A2W2;
constexpr size_t OFF_G1W2 = OFF_G1W1 + SZ_GW1;
constexpr size_t OFF_G2W1 = OFF_G1W2 + SZ_GW2;
constexpr size_t OFF_G2W2 = OFF_G2W1 + SZ_GW1;
constexpr size_t OFF_OW1  = OFF_G2W2 + SZ_GW2;

// ---------------- dynamic LDS layout ----------------
constexpr int L_XHL  = 16 * cKL;    // bf16
constexpr int L_XHA  = 16 * cKA;    // bf16
constexpr int L_CS   = 16 * cCSW;   // bf16
constexpr int L_FIN  = 16 * 768;    // bf16
constexpr int L_HIDB = 16 * 256;    // bf16
constexpr int L_CL   = 16 * cHL;    // f32
constexpr int L_CA   = 16 * cHA;    // f32
constexpr int L_LOG  = 16 * 1280;   // f32
constexpr int L_128  = 16 * 128;    // f32 (CHAT / G1 / G2)
constexpr size_t SMEM_BYTES =
    (size_t)(L_XHL + L_XHA + L_CS + L_FIN + L_HIDB) * 2 +
    (size_t)(L_CL + L_CA + L_LOG + 3 * L_128 + 64) * 4;

// ======================= weight repack (f32 -> bf16, WMMA-B order) ==========
__global__ void pack_weight(const float* __restrict__ W1, const float* __restrict__ W2,
                            __bf16* __restrict__ dst, int N, int K1, int K1pad, int K2, int Kp) {
  int ktiles = Kp >> 5;
  long total = (long)(N >> 4) * ktiles * 512;
  for (long e = (long)blockIdx.x * blockDim.x + threadIdx.x; e < total;
       e += (long)gridDim.x * blockDim.x) {
    int  j    = (int)(e & 511);
    long tile = e >> 9;
    int  ti_k = (int)(tile % ktiles);
    int  ti_n = (int)(tile / ktiles);
    int  l = j >> 4, s = j & 15;
    int  nloc = l & 15;
    int  half = (l >> 4) & 1;
    int  kloc = (s < 8) ? ((half ? 8 : 0) + s) : ((half ? 24 : 16) + (s - 8));
    int  n = ti_n * 16 + nloc, k = ti_k * 32 + kloc;
    float v = 0.f;
    if (k < K1)                                   v = W1[(long)n * K1 + k];
    else if (W2 && k >= K1pad && k < K1pad + K2)  v = W2[(long)n * K2 + (k - K1pad)];
    dst[e] = (__bf16)v;
  }
}

// ======================= WMMA helpers =======================================
__device__ __forceinline__ v16bf load_a_frag(const __bf16* base, int stride, int k0) {
  int lane = threadIdx.x & 31;
  int row  = lane & 15;
  int kk   = k0 + ((lane & 16) ? 8 : 0);
  const v8bf* p0 = (const v8bf*)(base + row * stride + kk);
  const v8bf* p1 = (const v8bf*)(base + row * stride + kk + 16);
  v8bf lo = *p0, hi = *p1;
  v16bf a;
#pragma unroll
  for (int i = 0; i < 8; ++i) { a[i] = lo[i]; a[i + 8] = hi[i]; }
  return a;
}

__device__ __forceinline__ v16bf load_b_frag(const __bf16* tile) {
  int lane = threadIdx.x & 31;
  const v8bf* p = (const v8bf*)(tile + lane * 16);
  v8bf lo = p[0], hi = p[1];
  v16bf b;
#pragma unroll
  for (int i = 0; i < 8; ++i) { b[i] = lo[i]; b[i + 8] = hi[i]; }
  return b;
}

__device__ __forceinline__ v8f wmma_bf16(v16bf a, v16bf b, v8f c) {
  return __builtin_amdgcn_wmma_f32_16x16x32_bf16(false, a, false, b, (short)0, c, false, false);
}

// D = A[16,K] * Wp(ntile)[K,16]  (accumulate over ktiles)
__device__ __forceinline__ v8f gemm_row(const __bf16* a_base, int a_stride,
                                        const __bf16* wp, int ktiles, int ntile) {
  v8f acc = {0.f, 0.f, 0.f, 0.f, 0.f, 0.f, 0.f, 0.f};
  const __bf16* bt = wp + (size_t)ntile * ktiles * 512;
  for (int kt = 0; kt < ktiles; ++kt) {
    if (kt + 1 < ktiles) __builtin_prefetch(bt + (size_t)(kt + 1) * 512, 0, 1);
    v16bf a = load_a_frag(a_base, a_stride, kt * 32);
    v16bf b = load_b_frag(bt + (size_t)kt * 512);
    acc = wmma_bf16(a, b, acc);
  }
  return acc;
}

__device__ __forceinline__ float sigf(float x) { return 1.f / (1.f + __expf(-x)); }

// acc (C/D layout: vgpr r -> M = r or 8+r; lane&15 -> N) -> LDS, with bias+act
__device__ __forceinline__ void store_f32(float* dst, int stride, int col0, v8f acc,
                                          const float* __restrict__ bias, int act) {
  int lane = threadIdx.x & 31;
  int n = lane & 15, mb = (lane & 16) ? 8 : 0;
  float b = bias ? bias[col0 + n] : 0.f;
#pragma unroll
  for (int r = 0; r < 8; ++r) {
    float v = acc[r] + b;
    if (act == 1) v = fmaxf(v, 0.f);
    else if (act == 2) v = sigf(v);
    else if (act == 3) v = tanhf(v);
    dst[(mb + r) * stride + col0 + n] = v;
  }
}

__device__ __forceinline__ void store_bf16(__bf16* dst, int stride, int col0, v8f acc,
                                           const float* __restrict__ bias, int act) {
  int lane = threadIdx.x & 31;
  int n = lane & 15, mb = (lane & 16) ? 8 : 0;
  float b = bias ? bias[col0 + n] : 0.f;
#pragma unroll
  for (int r = 0; r < 8; ++r) {
    float v = acc[r] + b;
    if (act == 1) v = fmaxf(v, 0.f);
    else if (act == 2) v = sigf(v);
    else if (act == 3) v = tanhf(v);
    dst[(mb + r) * stride + col0 + n] = (__bf16)v;
  }
}

// ======================= persistent recurrent kernel ========================
__global__ __launch_bounds__(NT, 1)
void async_lstm_main(const float* __restrict__ x_ling, const float* __restrict__ x_ac,
                     const int* __restrict__ seq_length,
                     const int* __restrict__ mask_ling, const int* __restrict__ mask_ac,
                     const int* __restrict__ lstm_masks,
                     const float* __restrict__ b_ih_l, const float* __restrict__ b_hh_l,
                     const float* __restrict__ b_ih_a, const float* __restrict__ b_hh_a,
                     const float* __restrict__ a1b1, const float* __restrict__ a1b2,
                     const float* __restrict__ a2b1, const float* __restrict__ a2b2,
                     const float* __restrict__ g1b1, const float* __restrict__ g1b2,
                     const float* __restrict__ g2b1, const float* __restrict__ g2b2,
                     const float* __restrict__ ob1, const float* __restrict__ out_w2,
                     const float* __restrict__ ob2,
                     const __bf16* __restrict__ wpk,
                     float* __restrict__ out) {
  extern __shared__ __align__(16) char smem_raw[];
  __bf16* XHL  = (__bf16*)smem_raw;        // [16][832]  x_l|h_l (bf16)
  __bf16* XHA  = XHL  + L_XHL;             // [16][224]  x_a|h_a
  __bf16* CS   = XHA  + L_XHA;             // [16][1408] cStar|mem
  __bf16* FIN  = CS   + L_CS;              // [16][768]  h_l|h_a|mem2
  __bf16* HIDB = FIN  + L_FIN;             // [16][256]  MLP hidden scratch
  float*  CL   = (float*)(HIDB + L_HIDB);  // [16][512]  c_l
  float*  CA   = CL   + L_CL;              // [16][128]  c_a
  float*  LOGI = CA   + L_CA;              // [16][1280] attention logits
  float*  CHAT = LOGI + L_LOG;             // [16][128]
  float*  G1o  = CHAT + L_128;             // [16][128]
  float*  G2o  = G1o  + L_128;             // [16][128]
  float*  PRL  = G2o  + L_128;             // [16] masks
  float*  PRA  = PRL + 16;
  float*  VALv = PRA + 16;

  const int tid  = threadIdx.x;
  const int wave = tid >> 5;
  const int lane = tid & 31;
  const int wgb  = blockIdx.x * 16;  // first batch row of this tile

  const __bf16* WL   = wpk + OFF_WL;
  const __bf16* WA   = wpk + OFF_WA;
  const __bf16* A1W1 = wpk + OFF_A1W1;
  const __bf16* A1W2 = wpk + OFF_A1W2;
  const __bf16* A2W1 = wpk + OFF_A2W1;
  const __bf16* A2W2 = wpk + OFF_A2W2;
  const __bf16* G1W1 = wpk + OFF_G1W1;
  const __bf16* G1W2 = wpk + OFF_G1W2;
  const __bf16* G2W1 = wpk + OFF_G2W1;
  const __bf16* G2W2 = wpk + OFF_G2W2;
  const __bf16* OW1  = wpk + OFF_OW1;

  // ---- init state ----
  for (int i = tid; i < L_CL; i += NT) CL[i] = 0.f;
  for (int i = tid; i < L_CA; i += NT) CA[i] = 0.f;
  for (int i = tid; i < 16 * cHL; i += NT) { int m = i >> 9, k = i & 511; XHL[m * cKL + 320 + k] = (__bf16)0.f; }
  for (int i = tid; i < 16 * cHA; i += NT) { int m = i >> 7, k = i & 127; XHA[m * cKA + 96  + k] = (__bf16)0.f; }
  for (int i = tid; i < 16 * cMEM; i += NT){ int m = i >> 7, k = i & 127; CS [m * cCSW + 1280 + k] = (__bf16)0.f; }
  __syncthreads();

  for (int t = 0; t < cT; ++t) {
    // ---- phase 0: snapshot prev c into cStar, load inputs + masks ----
    for (int i = tid; i < 16 * cHL; i += NT) { int m = i >> 9, k = i & 511; CS[m * cCSW + k] = (__bf16)CL[i]; }
    for (int i = tid; i < 16 * cHA; i += NT) { int m = i >> 7, k = i & 127; CS[m * cCSW + 512 + k] = (__bf16)CA[i]; }
    for (int i = tid; i < 16 * 320; i += NT) {
      int m = i / 320, k = i % 320;
      float v = (k < cDL) ? x_ling[((size_t)(wgb + m) * cT + t) * cDL + k] : 0.f;
      XHL[m * cKL + k] = (__bf16)v;
    }
    for (int i = tid; i < 16 * 96; i += NT) {
      int m = i / 96, k = i % 96;
      float v = (k < cDA) ? x_ac[((size_t)(wgb + m) * cT + t) * cDA + k] : 0.f;
      XHA[m * cKA + k] = (__bf16)v;
    }
    if (tid < 16) {
      int b = wgb + tid;
      bool valid = t < seq_length[b];
      PRL[tid]  = (valid && mask_ling[b * cT + t] > 0) ? 1.f : 0.f;
      PRA[tid]  = (valid && mask_ac[b * cT + t]   > 0) ? 1.f : 0.f;
      VALv[tid] = valid ? 1.f : 0.f;
    }
    __syncthreads();

    // ---- LSTM l : gates = [x|h] @ [Wih|Whh]^T, fused epilogue ----
    for (int hb = wave; hb < 32; hb += NWAVE) {
      v8f ai = gemm_row(XHL, cKL, WL, 26, hb);
      v8f af = gemm_row(XHL, cKL, WL, 26, 32 + hb);
      v8f ag = gemm_row(XHL, cKL, WL, 26, 64 + hb);
      v8f ao = gemm_row(XHL, cKL, WL, 26, 96 + hb);
      int n = lane & 15, mb = (lane & 16) ? 8 : 0;
      int col = hb * 16 + n;
      float bi = b_ih_l[col]        + b_hh_l[col];
      float bf = b_ih_l[512 + col]  + b_hh_l[512 + col];
      float bg = b_ih_l[1024 + col] + b_hh_l[1024 + col];
      float bo = b_ih_l[1536 + col] + b_hh_l[1536 + col];
#pragma unroll
      for (int r = 0; r < 8; ++r) {
        int m = mb + r;
        float ig = sigf(ai[r] + bi), fg = sigf(af[r] + bf);
        float gg = tanhf(ag[r] + bg), og = sigf(ao[r] + bo);
        float cp = CL[m * cHL + col];
        float c2 = fg * cp + ig * gg;
        float h2 = og * tanhf(c2);
        float hp = (float)XHL[m * cKL + 320 + col];
        float pres = PRL[m], val = VALv[m];
        float nc = pres > 0.5f ? c2 : (val > 0.5f ? cp : 0.f);
        float nh = pres > 0.5f ? h2 : (val > 0.5f ? hp : 0.f);
        CL[m * cHL + col] = nc;
        CS[m * cCSW + 640 + col] = (__bf16)nc;
        FIN[m * 768 + col] = (__bf16)nh;
      }
    }
    // ---- LSTM a ----
    for (int hb = wave; hb < 8; hb += NWAVE) {
      v8f ai = gemm_row(XHA, cKA, WA, 7, hb);
      v8f af = gemm_row(XHA, cKA, WA, 7, 8 + hb);
      v8f ag = gemm_row(XHA, cKA, WA, 7, 16 + hb);
      v8f ao = gemm_row(XHA, cKA, WA, 7, 24 + hb);
      int n = lane & 15, mb = (lane & 16) ? 8 : 0;
      int col = hb * 16 + n;
      float bi = b_ih_a[col]       + b_hh_a[col];
      float bf = b_ih_a[128 + col] + b_hh_a[128 + col];
      float bg = b_ih_a[256 + col] + b_hh_a[256 + col];
      float bo = b_ih_a[384 + col] + b_hh_a[384 + col];
#pragma unroll
      for (int r = 0; r < 8; ++r) {
        int m = mb + r;
        float ig = sigf(ai[r] + bi), fg = sigf(af[r] + bf);
        float gg = tanhf(ag[r] + bg), og = sigf(ao[r] + bo);
        float cp = CA[m * cHA + col];
        float c2 = fg * cp + ig * gg;
        float h2 = og * tanhf(c2);
        float hp = (float)XHA[m * cKA + 96 + col];
        float pres = PRA[m], val = VALv[m];
        float nc = pres > 0.5f ? c2 : (val > 0.5f ? cp : 0.f);
        float nh = pres > 0.5f ? h2 : (val > 0.5f ? hp : 0.f);
        CA[m * cHA + col] = nc;
        CS[m * cCSW + 1152 + col] = (__bf16)nc;
        FIN[m * 768 + 512 + col] = (__bf16)nh;
      }
    }
    __syncthreads();

    // ---- commit new h into GEMM input buffers ----
    for (int i = tid; i < 16 * cHL; i += NT) { int m = i >> 9, k = i & 511; XHL[m * cKL + 320 + k] = FIN[m * 768 + k]; }
    for (int i = tid; i < 16 * cHA; i += NT) { int m = i >> 7, k = i & 127; XHA[m * cKA + 96 + k] = FIN[m * 768 + 512 + k]; }
    __syncthreads();

    // ---- att1 hidden: relu(cStar @ w1^T + b1) -> HIDB[:, :128] ----
    for (int nt = wave; nt < 8; nt += NWAVE)
      store_bf16(HIDB, 256, nt * 16, gemm_row(CS, cCSW, A1W1, 40, nt), a1b1, 1);
    __syncthreads();
    // ---- att1 logits: HIDB @ w2^T + b2 -> LOGI[16][1280] ----
    for (int nt = wave; nt < 80; nt += NWAVE)
      store_f32(LOGI, 1280, nt * 16, gemm_row(HIDB, 256, A1W2, 4, nt), a1b2, 0);
    __syncthreads();
    // ---- softmax over 1280, attended = softmax * cStar (in place in CS) ----
    if (tid < 16) {
      int r = tid;
      float mx = -3.0e38f;
      for (int k = 0; k < 1280; ++k) mx = fmaxf(mx, LOGI[r * 1280 + k]);
      float s = 0.f;
      for (int k = 0; k < 1280; ++k) { float e = __expf(LOGI[r * 1280 + k] - mx); LOGI[r * 1280 + k] = e; s += e; }
      float inv = 1.f / s;
      for (int k = 0; k < 1280; ++k) {
        float att = LOGI[r * 1280 + k] * inv;
        float cs  = (float)CS[r * cCSW + k];
        CS[r * cCSW + k] = (__bf16)(att * cs);
      }
    }
    __syncthreads();

    // ---- att2 hidden: relu(attended @ w1^T + b1) -> HIDB[:, :256] ----
    for (int nt = wave; nt < 16; nt += NWAVE)
      store_bf16(HIDB, 256, nt * 16, gemm_row(CS, cCSW, A2W1, 40, nt), a2b1, 1);
    __syncthreads();
    // ---- cHat = tanh(hidden @ w2^T + b2) ----
    for (int nt = wave; nt < 8; nt += NWAVE)
      store_f32(CHAT, 128, nt * 16, gemm_row(HIDB, 256, A2W2, 8, nt), a2b2, 3);
    __syncthreads();

    // ---- gamma hidden layers (both = [attended|mem] = CS[:, :1408]) ----
    for (int nt = wave; nt < 4; nt += NWAVE) {
      store_bf16(HIDB, 256, nt * 16,      gemm_row(CS, cCSW, G1W1, 44, nt), g1b1, 1);
      store_bf16(HIDB, 256, 64 + nt * 16, gemm_row(CS, cCSW, G2W1, 44, nt), g2b1, 1);
    }
    __syncthreads();
    // ---- gamma outputs ----
    for (int nt = wave; nt < 8; nt += NWAVE) {
      store_f32(G1o, 128, nt * 16, gemm_row(HIDB,      256, G1W2, 2, nt), g1b2, 2);
      store_f32(G2o, 128, nt * 16, gemm_row(HIDB + 64, 256, G2W2, 2, nt), g2b2, 2);
    }
    __syncthreads();

    // ---- mem update: mem2 = g1*mem + g2*cHat ----
    for (int i = tid; i < 16 * cMEM; i += NT) {
      int m = i >> 7, k = i & 127;
      float mprev = (float)CS[m * cCSW + 1280 + k];
      float m2 = G1o[i] * mprev + G2o[i] * CHAT[i];
      CS[m * cCSW + 1280 + k] = (__bf16)m2;
      FIN[m * 768 + 640 + k] = (__bf16)m2;
    }
    __syncthreads();

    // ---- output projection: relu(FIN @ ow1^T + b1) -> HIDB[:, :64] ----
    for (int nt = wave; nt < 4; nt += NWAVE)
      store_bf16(HIDB, 256, nt * 16, gemm_row(FIN, 768, OW1, 24, nt), ob1, 1);
    __syncthreads();
    // ---- final scalar + mask -> d_out ----
    if (tid < 16) {
      int r = tid, b = wgb + r;
      float s = ob2[0];
      for (int j = 0; j < 64; ++j) s += (float)HIDB[r * 256 + j] * out_w2[j];
      float mk = (float)lstm_masks[b * cT + t];
      out[(size_t)b * cT + t] = s * mk;
    }
    __syncthreads();
  }
}

// ======================= host launch ========================================
extern "C" void kernel_launch(void* const* d_in, const int* in_sizes, int n_in,
                              void* d_out, int out_size, void* d_ws, size_t ws_size,
                              hipStream_t stream) {
  (void)in_sizes; (void)n_in; (void)out_size; (void)ws_size;
  const float* x_ling  = (const float*)d_in[0];
  const float* x_ac    = (const float*)d_in[1];
  const int*   seq_len = (const int*)d_in[2];
  const int*   m_ling  = (const int*)d_in[3];
  const int*   m_ac    = (const int*)d_in[4];
  const int*   l_masks = (const int*)d_in[5];
  const float* W_ih_l  = (const float*)d_in[6];
  const float* W_hh_l  = (const float*)d_in[7];
  const float* b_ih_l  = (const float*)d_in[8];
  const float* b_hh_l  = (const float*)d_in[9];
  const float* W_ih_a  = (const float*)d_in[10];
  const float* W_hh_a  = (const float*)d_in[11];
  const float* b_ih_a  = (const float*)d_in[12];
  const float* b_hh_a  = (const float*)d_in[13];
  const float* a1w1 = (const float*)d_in[14]; const float* a1b1 = (const float*)d_in[15];
  const float* a1w2 = (const float*)d_in[16]; const float* a1b2 = (const float*)d_in[17];
  const float* a2w1 = (const float*)d_in[18]; const float* a2b1 = (const float*)d_in[19];
  const float* a2w2 = (const float*)d_in[20]; const float* a2b2 = (const float*)d_in[21];
  const float* g1w1 = (const float*)d_in[22]; const float* g1b1 = (const float*)d_in[23];
  const float* g1w2 = (const float*)d_in[24]; const float* g1b2 = (const float*)d_in[25];
  const float* g2w1 = (const float*)d_in[26]; const float* g2b1 = (const float*)d_in[27];
  const float* g2w2 = (const float*)d_in[28]; const float* g2b2 = (const float*)d_in[29];
  const float* ow1  = (const float*)d_in[30]; const float* ob1  = (const float*)d_in[31];
  const float* ow2  = (const float*)d_in[32]; const float* ob2  = (const float*)d_in[33];

  __bf16* ws = (__bf16*)d_ws;

  auto pk = [&](const float* W1, const float* W2, size_t off,
                int N, int K1, int K1pad, int K2, int Kp) {
    size_t total = (size_t)(N / 16) * (Kp / 32) * 512;
    int blocks = (int)((total + 255) / 256);
    if (blocks > 4096) blocks = 4096;
    pack_weight<<<blocks, 256, 0, stream>>>(W1, W2, ws + off, N, K1, K1pad, K2, Kp);
  };

  pk(W_ih_l, W_hh_l, OFF_WL,   2048, 300, 320, 512, 832);
  pk(W_ih_a, W_hh_a, OFF_WA,   512,  74,  96,  128, 224);
  pk(a1w1, nullptr,  OFF_A1W1, 128,  1280, 1280, 0, 1280);
  pk(a1w2, nullptr,  OFF_A1W2, 1280, 128,  128,  0, 128);
  pk(a2w1, nullptr,  OFF_A2W1, 256,  1280, 1280, 0, 1280);
  pk(a2w2, nullptr,  OFF_A2W2, 128,  256,  256,  0, 256);
  pk(g1w1, nullptr,  OFF_G1W1, 64,   1408, 1408, 0, 1408);
  pk(g1w2, nullptr,  OFF_G1W2, 128,  64,   64,   0, 64);
  pk(g2w1, nullptr,  OFF_G2W1, 64,   1408, 1408, 0, 1408);
  pk(g2w2, nullptr,  OFF_G2W2, 128,  64,   64,   0, 64);
  pk(ow1,  nullptr,  OFF_OW1,  64,   768,  768,  0, 768);

  hipFuncSetAttribute(reinterpret_cast<const void*>(async_lstm_main),
                      hipFuncAttributeMaxDynamicSharedMemorySize, (int)SMEM_BYTES);

  async_lstm_main<<<cB / 16, NT, SMEM_BYTES, stream>>>(
      x_ling, x_ac, seq_len, m_ling, m_ac, l_masks,
      b_ih_l, b_hh_l, b_ih_a, b_hh_a,
      a1b1, a1b2, a2b1, a2b2, g1b1, g1b2, g2b1, g2b2,
      ob1, ow2, ob2, ws, (float*)d_out);
}
